// MultiheadAttention_77567109366503
// MI455X (gfx1250) — matrix operand order repaired and checked by reading, not compiled, not run on hardware
//
#include <hip/hip_runtime.h>
#include <hip/hip_bf16.h>

typedef __attribute__((ext_vector_type(16))) _Float16 v16h;
typedef __attribute__((ext_vector_type(8)))  _Float16 v8h;
typedef __attribute__((ext_vector_type(4)))  _Float16 v4h;
typedef __attribute__((ext_vector_type(8)))  float    v8f;

// ---- helpers -------------------------------------------------------------

__device__ inline v16h cat8(v8h a, v8h b) {
  return __builtin_shufflevector(a, b, 0,1,2,3,4,5,6,7,8,9,10,11,12,13,14,15);
}

__device__ inline v8f wmma16(v16h a, v16h b, v8f c) {
  return __builtin_amdgcn_wmma_f32_16x16x32_f16(false, a, false, b, (short)0, c,
                                                false, false);
}

// D = 1024, B = 4, S = 1024, H = 16, dk = 64

// ---- kernel 0: fp32 -> f16 bulk convert (n4 = element count / 4) ---------
__global__ __launch_bounds__(256) void cvt_kernel(const float* __restrict__ in,
                                                  _Float16* __restrict__ out,
                                                  int n4) {
  const int stride = gridDim.x * 256;
  for (int i = blockIdx.x * 256 + threadIdx.x; i < n4; i += stride) {
    const float4 v = ((const float4*)in)[i];
    v4h o;
    o[0] = (_Float16)v.x; o[1] = (_Float16)v.y;
    o[2] = (_Float16)v.z; o[3] = (_Float16)v.w;
    ((v4h*)out)[i] = o;
  }
}

// ---- kernel 1: Y = X @ W^T + bias (all-f16 operands, f32 accum) ----------
// Each wave computes a 16x64 tile (4 accumulators, A-fragment reused 4x).
// mode 0: out[((b*16+h)*1024 + s)*64 + d]   (q, k layout)
// mode 1: out[((b*16+h)*64 + d)*1024 + s]   (v transposed layout)
__global__ __launch_bounds__(256) void proj_kernel(
    const _Float16* __restrict__ X, const _Float16* __restrict__ W,
    const float* __restrict__ bias, _Float16* __restrict__ out, int mode) {
  const int tid  = threadIdx.x;
  const int wave = tid >> 5, lane = tid & 31;
  const int m0 = blockIdx.x * 16;
  const int n0 = blockIdx.y * 512 + wave * 64;
  const int Mr  = lane & 15;
  const int hi  = lane >> 4;
  const int kb8 = hi * 8, kb16 = hi * 16;
  const _Float16* arow = X + (m0 + Mr) * 1024;

  v8f acc[4] = {};
  for (int k0 = 0; k0 < 1024; k0 += 32) {
    const _Float16* pa = arow + k0 + kb16;
    __builtin_prefetch(pa + 64, 0, 1);
    v16h afr = *(const v16h*)pa;
    for (int t = 0; t < 4; ++t) {
      const _Float16* pb = W + (n0 + t * 16 + Mr) * 1024 + k0 + kb16;
      acc[t] = wmma16(afr, *(const v16h*)pb, acc[t]);
    }
  }

  const int bb = m0 >> 10;          // batch (S = 1024, tile never crosses)
  const int sbase = m0 & 1023;
  for (int t = 0; t < 4; ++t) {
    const int col = n0 + t * 16 + Mr;
    const float bv = bias[col];
    const int h = col >> 6, d = col & 63;
    for (int r = 0; r < 8; ++r) {
      const int s = sbase + r + hi * 8;
      const float val = acc[t][r] + bv;
      int idx;
      if (mode == 0) idx = ((bb * 16 + h) * 1024 + s) * 64 + d;
      else           idx = ((bb * 16 + h) * 64 + d) * 1024 + s;
      out[idx] = (_Float16)val;
    }
  }
}

// ---- kernel 2: attention with double softmax -----------------------------
// q,k: [B,H,S,64] f16; vt: [B,H,64,S] f16; ctx: [B,S,D] f16
__global__ __launch_bounds__(256) void attn_kernel(
    const _Float16* __restrict__ q, const _Float16* __restrict__ k,
    const _Float16* __restrict__ vt, _Float16* __restrict__ ctx) {
  __shared__ __align__(16) _Float16 sc[16 * 1024];  // score strip, fp16
  __shared__ float red[256];                        // softmax reductions
  __shared__ float pred[4 * 256];                   // K-split partials

  const int tid  = threadIdx.x;
  const int wave = tid >> 5, lane = tid & 31;
  const int bt = blockIdx.z, h = blockIdx.y;
  const int s0 = blockIdx.x * 16;
  const int Mr  = lane & 15;
  const int hi  = lane >> 4;
  const int kb8  = hi * 8, kb16 = hi * 16;

  const _Float16* qh = q  + ((bt * 16 + h) * 1024) * 64;
  const _Float16* kh = k  + ((bt * 16 + h) * 1024) * 64;
  const _Float16* vh = vt + ((bt * 16 + h) * 64) * 1024;

  // Q row-tile fragments for the two dk halves (reused across 8 col tiles)
  v16h qa[2];
  for (int kk = 0; kk < 2; ++kk) {
    const _Float16* p = qh + (s0 + Mr) * 64 + kk * 32 + kb16;
    qa[kk] = *(const v16h*)p;
  }

  // Phase 1: scores = (q @ k^T) / sqrt(64)
  for (int j = wave * 8; j < wave * 8 + 8; ++j) {
    v8f acc = {};
    for (int kk = 0; kk < 2; ++kk) {
      const _Float16* p = kh + (j * 16 + Mr) * 64 + kk * 32 + kb16;
      acc = wmma16(qa[kk], *(const v16h*)p, acc);
    }
    for (int r = 0; r < 8; ++r)
      sc[(r + hi * 8) * 1024 + j * 16 + Mr] = (_Float16)(acc[r] * 0.125f);
  }
  __syncthreads();

  // Phase 2: softmax applied twice (reference quirk)
  {
    const int row = tid >> 4, sub = tid & 15;
    for (int pass = 0; pass < 2; ++pass) {
      float pm = -1e30f;
      for (int i = 0; i < 64; ++i)
        pm = fmaxf(pm, (float)sc[row * 1024 + sub + i * 16]);
      red[tid] = pm; __syncthreads();
      float rm = -1e30f;
      for (int j = 0; j < 16; ++j) rm = fmaxf(rm, red[row * 16 + j]);
      __syncthreads();
      float ps = 0.f;
      for (int i = 0; i < 64; ++i) {
        const int idx = row * 1024 + sub + i * 16;
        const float e = __expf((float)sc[idx] - rm);
        sc[idx] = (_Float16)e;
        ps += e;
      }
      red[tid] = ps; __syncthreads();
      float rs = 0.f;
      for (int j = 0; j < 16; ++j) rs += red[row * 16 + j];
      __syncthreads();
      const float inv = 1.0f / rs;
      for (int i = 0; i < 64; ++i) {
        const int idx = row * 1024 + sub + i * 16;
        sc[idx] = (_Float16)((float)sc[idx] * inv);
      }
      __syncthreads();
    }
  }

  // Phase 3: ctx = attn @ v  (4 N-tiles x 2 K-halves across 8 waves)
  const int ntile = wave & 3, khalf = wave >> 2;
  v8f cacc = {};
  for (int ks = 0; ks < 16; ++ks) {
    const int k0 = khalf * 512 + ks * 32;
    const _Float16* pa = &sc[Mr * 1024 + k0 + kb16];
    v16h afr = cat8(*(const v8h*)pa, *(const v8h*)(pa + 8));
    const _Float16* pb = vh + (ntile * 16 + Mr) * 1024 + k0 + kb16;
    cacc = wmma16(afr, *(const v16h*)pb, cacc);
  }
  if (wave >= 4)
    for (int r = 0; r < 8; ++r) pred[(wave - 4) * 256 + r * 32 + lane] = cacc[r];
  __syncthreads();
  if (wave < 4) {
    for (int r = 0; r < 8; ++r) cacc[r] += pred[wave * 256 + r * 32 + lane];
    const int dcol = h * 64 + ntile * 16 + Mr;
    for (int r = 0; r < 8; ++r) {
      const int s = s0 + r + hi * 8;
      ctx[(bt * 1024 + s) * 1024 + dcol] = (_Float16)cacc[r];
    }
  }
}

// ---- kernel 3: out = LN(ctx @ Wo^T + bo + Qres) --------------------------
__global__ __launch_bounds__(256) void outproj_ln_kernel(
    const _Float16* __restrict__ ctx, const _Float16* __restrict__ Wo,
    const float* __restrict__ bo, const float* __restrict__ Qin,
    const float* __restrict__ g, const float* __restrict__ bln,
    float* __restrict__ out) {
  __shared__ float ssum[16];
  __shared__ float ssq[16];
  const int tid  = threadIdx.x;
  const int wave = tid >> 5, lane = tid & 31;
  const int m0 = blockIdx.x * 16;
  const int Mr  = lane & 15;
  const int hi  = lane >> 4;
  const int kb8 = hi * 8, kb16 = hi * 16;
  const _Float16* arow = ctx + (m0 + Mr) * 1024;

  v8f acc[8] = {};
  for (int k0 = 0; k0 < 1024; k0 += 32) {
    const _Float16* pa = arow + k0 + kb16;
    __builtin_prefetch(pa + 64, 0, 1);
    v16h afr = *(const v16h*)pa;
    for (int t = 0; t < 8; ++t) {
      const int n = (wave * 8 + t) * 16 + Mr;
      const _Float16* pb = Wo + n * 1024 + k0 + kb16;
      acc[t] = wmma16(afr, *(const v16h*)pb, acc[t]);
    }
  }

  if (tid < 16) { ssum[tid] = 0.f; ssq[tid] = 0.f; }
  __syncthreads();

  // bias + residual, accumulate LN statistics from registers
  float ls[8] = {0,0,0,0,0,0,0,0};
  float lq[8] = {0,0,0,0,0,0,0,0};
  for (int t = 0; t < 8; ++t) {
    const int col = (wave * 8 + t) * 16 + Mr;
    const float bb = bo[col];
    for (int r = 0; r < 8; ++r) {
      const int m = m0 + r + hi * 8;
      const float x = acc[t][r] + bb + Qin[m * 1024 + col];
      acc[t][r] = x;
      ls[r] += x;
      lq[r] += x * x;
    }
  }
  for (int r = 0; r < 8; ++r) {
    atomicAdd(&ssum[r + hi * 8], ls[r]);
    atomicAdd(&ssq[r + hi * 8], lq[r]);
  }
  __syncthreads();

  for (int t = 0; t < 8; ++t) {
    const int col = (wave * 8 + t) * 16 + Mr;
    const float gg = g[col], b2 = bln[col];
    for (int r = 0; r < 8; ++r) {
      const int row = r + hi * 8;
      const int m = m0 + row;
      const float mu  = ssum[row] * (1.0f / 1024.0f);
      const float var = ssq[row] * (1.0f / 1024.0f) - mu * mu;
      out[m * 1024 + col] = (acc[t][r] - mu) * rsqrtf(var + 1e-5f) * gg + b2;
    }
  }
}

// ---- launcher ------------------------------------------------------------

extern "C" void kernel_launch(void* const* d_in, const int* in_sizes, int n_in,
                              void* d_out, int out_size, void* d_ws, size_t ws_size,
                              hipStream_t stream) {
  const float* Q   = (const float*)d_in[0];
  const float* K   = (const float*)d_in[1];
  const float* V   = (const float*)d_in[2];
  const float* Wq  = (const float*)d_in[3];
  const float* bq  = (const float*)d_in[4];
  const float* Wk  = (const float*)d_in[5];
  const float* bk  = (const float*)d_in[6];
  const float* Wv  = (const float*)d_in[7];
  const float* bv  = (const float*)d_in[8];
  const float* Wo  = (const float*)d_in[9];
  const float* bo  = (const float*)d_in[10];
  const float* lng = (const float*)d_in[11];
  const float* lnb = (const float*)d_in[12];
  float* out = (float*)d_out;

  const size_t MB = 1u << 20;
  char* ws = (char*)d_ws;
  _Float16* Qh  = (_Float16*)(ws);               // 8 MB  (f16 copy of Q)
  _Float16* Kh  = (_Float16*)(ws + 8  * MB);     // 8 MB
  _Float16* Vh  = (_Float16*)(ws + 16 * MB);     // 8 MB
  _Float16* Wqh = (_Float16*)(ws + 24 * MB);     // 2 MB
  _Float16* Wkh = (_Float16*)(ws + 26 * MB);     // 2 MB
  _Float16* Wvh = (_Float16*)(ws + 28 * MB);     // 2 MB
  _Float16* Woh = (_Float16*)(ws + 30 * MB);     // 2 MB
  _Float16* qh  = (_Float16*)(ws + 32 * MB);     // 8 MB  [B,H,S,dk]
  _Float16* kh  = (_Float16*)(ws + 40 * MB);     // 8 MB  [B,H,S,dk]
  _Float16* vt  = (_Float16*)(ws + 48 * MB);     // 8 MB  [B,H,dk,S]
  _Float16* ctx = (_Float16*)(ws + 56 * MB);     // 8 MB  [B,S,D]

  const int nX4 = (4 * 1024 * 1024) / 4;   // B*S*D / 4
  const int nW4 = (1024 * 1024) / 4;       // D*D / 4
  cvt_kernel<<<512, 256, 0, stream>>>(Q,  Qh,  nX4);
  cvt_kernel<<<512, 256, 0, stream>>>(K,  Kh,  nX4);
  cvt_kernel<<<512, 256, 0, stream>>>(V,  Vh,  nX4);
  cvt_kernel<<<256, 256, 0, stream>>>(Wq, Wqh, nW4);
  cvt_kernel<<<256, 256, 0, stream>>>(Wk, Wkh, nW4);
  cvt_kernel<<<256, 256, 0, stream>>>(Wv, Wvh, nW4);
  cvt_kernel<<<256, 256, 0, stream>>>(Wo, Woh, nW4);

  dim3 gp(256, 2);
  proj_kernel<<<gp, 256, 0, stream>>>(Qh, Wqh, bq, qh, 0);
  proj_kernel<<<gp, 256, 0, stream>>>(Kh, Wkh, bk, kh, 0);
  proj_kernel<<<gp, 256, 0, stream>>>(Vh, Wvh, bv, vt, 1);
  attn_kernel<<<dim3(64, 16, 4), 256, 0, stream>>>(qh, kh, vt, ctx);
  outproj_ln_kernel<<<256, 256, 0, stream>>>(ctx, Woh, bo, Q, lng, lnb, out);
}
